// SegRNNBlock_36180804502071
// MI455X (gfx1250) — matrix operand order repaired and checked by reading, not compiled, not run on hardware
//
#include <hip/hip_runtime.h>
#include <hip/hip_bf16.h>
#include <math.h>

// Problem constants (from reference)
#define BATCH 1024
#define KSEG  160          // K = IN_F / WS
#define WS    12           // segment width == GRU hidden size == OUT_WS
#define GATE  (3*WS)       // 36
#define DHID  1024
#define OUTF  1920         // == KSEG*WS
#define MT    32           // decoder batch rows per wave (2 WMMA A-tiles)

typedef __attribute__((ext_vector_type(16))) __bf16 v16bf;
typedef __attribute__((ext_vector_type(8)))  float  v8f;

union BF16x16 {
    v16bf v;
    unsigned short u[16];
    uint4 q[2];            // two 16-byte chunks
};

// round-to-nearest-even f32 -> bf16 bits (cold paths: weight packing, A build)
__device__ __forceinline__ unsigned short f2bf(float f) {
    unsigned int u = __float_as_uint(f);
    unsigned int r = u + 0x7FFFu + ((u >> 16) & 1u);
    return (unsigned short)(r >> 16);
}
// truncating f32 -> bf16 bits (hot path: lowers to ds_store_b16_d16_hi, zero VALU)
__device__ __forceinline__ unsigned short f2bf_trunc(float f) {
    return (unsigned short)(__float_as_uint(f) >> 16);
}

// ---------------------------------------------------------------------------
// Kernel 1: encoder  xs[b,k,o] = relu( sum_i x[b,k,i] * enc_W[k,i,o] + enc_b[k,o] )
// ---------------------------------------------------------------------------
__global__ void enc_kernel(const float* __restrict__ x, const float* __restrict__ W,
                           const float* __restrict__ b, float* __restrict__ xs) {
    int idx = blockIdx.x * blockDim.x + threadIdx.x;
    if (idx >= BATCH * KSEG * WS) return;
    int o  = idx % WS;
    int k  = (idx / WS) % KSEG;
    int bb = idx / (WS * KSEG);
    const float* xr = x + (size_t)(bb * KSEG + k) * WS;
    const float* wr = W + (size_t)k * WS * WS + o;   // stride WS over i
    float acc = b[k * WS + o];
    #pragma unroll
    for (int i = 0; i < WS; ++i) acc += xr[i] * wr[i * WS];
    xs[idx] = fmaxf(acc, 0.0f);
}

// ---------------------------------------------------------------------------
// Kernel 2: positional + channel encoding table, pe[t,d]
//   pos_enc(160,12)[t,d] + sin(t)   (arange(12)//160 == 0 -> ch_pe[:,0] = sin(t))
// ---------------------------------------------------------------------------
__global__ void pe_kernel(float* __restrict__ pe) {
    int idx = blockIdx.x * blockDim.x + threadIdx.x;
    if (idx >= KSEG * WS) return;
    int d = idx % WS, t = idx / WS;
    float div = __expf((float)(d & ~1) * (-logf(10000.0f) / (float)WS));
    float ang = (float)t * div;
    float v = (d & 1) ? __cosf(ang) : __sinf(ang);
    pe[idx] = v + __sinf((float)t);
}

// ---------------------------------------------------------------------------
// Kernel 3a: pack dec_W1 (K,12,1024) + dec_b1 into bf16 WMMA B-layout.
// B 32x16 (16-bit): per lane (N = lane&15), element e holds K = (e<8 ? e : 16+e-8) + (lane>>4)*8.
// K rows 0..11 = W1; K row 12 = b1 (bias, paired with ones-row in A); rest 0.
// One thread emits one lane's 16 shorts (32 contiguous bytes).
// ---------------------------------------------------------------------------
__global__ void pack_w1_kernel(const float* __restrict__ W1, const float* __restrict__ b1,
                               unsigned short* __restrict__ W1bf) {
    int idx = blockIdx.x * blockDim.x + threadIdx.x;   // ((kk*64)+n)*32 + lane
    if (idx >= KSEG * 64 * 32) return;
    int lane = idx & 31, n = (idx >> 5) & 63, kk = idx >> 11;
    int M = lane & 15, hi = lane >> 4;
    const float* W1k = W1 + (size_t)kk * WS * DHID;
    float bias = b1[(size_t)kk * DHID + n * 16 + M];
    BF16x16 o;
    #pragma unroll
    for (int e = 0; e < 16; ++e) {
        int ke = ((e < 8) ? e : (16 + e - 8)) + hi * 8;
        unsigned short val = 0;
        if (ke < WS)       val = f2bf(W1k[(size_t)ke * DHID + n * 16 + M]);
        else if (ke == WS) val = f2bf(bias);
        o.u[e] = val;
    }
    uint4* dst = (uint4*)(W1bf + (size_t)idx * 16);
    dst[0] = o.q[0];
    dst[1] = o.q[1];
}

// ---------------------------------------------------------------------------
// Kernel 3b: pack dec_W2 (K,1024,12) into bf16 WMMA B-layout (zero-padded N 12->16).
// ---------------------------------------------------------------------------
__global__ void pack_w2_kernel(const float* __restrict__ W2,
                               unsigned short* __restrict__ W2bf) {
    int idx = blockIdx.x * blockDim.x + threadIdx.x;   // ((kk*32)+c2)*32 + lane
    if (idx >= KSEG * 32 * 32) return;
    int lane = idx & 31, c2 = (idx >> 5) & 31, kk = idx >> 10;
    int M = lane & 15, hi = lane >> 4;
    const float* W2k = W2 + (size_t)kk * DHID * WS;
    BF16x16 o;
    #pragma unroll
    for (int e = 0; e < 16; ++e) {
        int kb = c2 * 32 + ((e < 8) ? e : (16 + e - 8)) + hi * 8;
        o.u[e] = (M < WS) ? f2bf(W2k[(size_t)kb * WS + M]) : (unsigned short)0;
    }
    uint4* dst = (uint4*)(W2bf + (size_t)idx * 16);
    dst[0] = o.q[0];
    dst[1] = o.q[1];
}

// ---------------------------------------------------------------------------
// Kernel 4: both GRU passes fused. Block = 16 batch rows x 16 lanes (12 active).
// ---------------------------------------------------------------------------
__global__ void __launch_bounds__(256) gru_kernel(
    const float* __restrict__ xs, const float* __restrict__ pe,
    const float* __restrict__ Wih, const float* __restrict__ Whh,
    const float* __restrict__ bih, const float* __restrict__ bhh,
    float* __restrict__ out) {
    __shared__ float sWih[GATE * WS], sWhh[GATE * WS];
    __shared__ float sbi[GATE], sbh[GATE];
    __shared__ float sPe[KSEG * WS];
    __shared__ float sh[16][WS + 1];
    __shared__ float sx[2][16][WS + 1];

    int tid = threadIdx.x;
    for (int i = tid; i < GATE * WS; i += 256) { sWih[i] = Wih[i]; sWhh[i] = Whh[i]; }
    if (tid < GATE) { sbi[tid] = bih[tid]; sbh[tid] = bhh[tid]; }
    for (int i = tid; i < KSEG * WS; i += 256) sPe[i] = pe[i];

    int j = tid & 15, g = tid >> 4;
    int b = blockIdx.x * 16 + g;
    const float* xrow = xs + (size_t)b * KSEG * WS;
    if (j < WS) { sh[g][j] = 0.0f; sx[0][g][j] = xrow[j]; }   // preload step 0
    __syncthreads();

    for (int s = 0; s < 2 * KSEG; ++s) {
        int t  = (s < KSEG) ? s : s - KSEG;
        int cb = s & 1, nb = cb ^ 1;
        if (j < WS && s + 1 < 2 * KSEG) {      // stage next step's input
            int sn = s + 1;
            int tn = (sn < KSEG) ? sn : sn - KSEG;
            float xv = xrow[tn * WS + j];
            if (sn >= KSEG) xv += sPe[tn * WS + j];
            sx[nb][g][j] = xv;
        }
        float hnew = 0.0f;
        if (j < WS) {
            float ir = sbi[j], iz = sbi[j + WS], in_ = sbi[j + 2 * WS];
            float hr = sbh[j], hz = sbh[j + WS], hn  = sbh[j + 2 * WS];
            #pragma unroll
            for (int i = 0; i < WS; ++i) {
                float xv = sx[cb][g][i], hv = sh[g][i];
                ir  += sWih[j * WS + i] * xv;
                iz  += sWih[(j + WS) * WS + i] * xv;
                in_ += sWih[(j + 2 * WS) * WS + i] * xv;
                hr  += sWhh[j * WS + i] * hv;
                hz  += sWhh[(j + WS) * WS + i] * hv;
                hn  += sWhh[(j + 2 * WS) * WS + i] * hv;
            }
            float r = 1.0f / (1.0f + __expf(-(ir + hr)));
            float z = 1.0f / (1.0f + __expf(-(iz + hz)));
            float n = tanhf(in_ + r * hn);
            hnew = (1.0f - z) * n + z * sh[g][j];
        }
        __syncthreads();
        if (j < WS) {
            sh[g][j] = hnew;
            if (s >= KSEG) out[(size_t)b * OUTF + t * WS + j] = hnew;  // ys staged in d_out
        }
        __syncthreads();
    }
}

// ---------------------------------------------------------------------------
// Kernel 5: decoder, bf16 WMMA, f32 accumulate. One wave per (k, 32-row tile):
// two 16-row A-tiles share every loaded B operand (loads:wmma = 1:1).
//   hid(32x1024) = [ysTile | 1](32x13 padded to 32) @ [W1;b1](13x1024)  (bias folded)
//   res(32x12)   = hid @ W2[k](1024x12) + b2[k]
// A 16x32 (16-bit): row = lane&15; element e -> K = (e<8 ? e : 16+e-8) + (lane>>4)*8
// C/D 16x16 f32:    element r -> row r + 8*(lane>>4), col = lane&15
// ---------------------------------------------------------------------------
__global__ void __launch_bounds__(32) dec_kernel(
    const unsigned short* __restrict__ W1bf, const unsigned short* __restrict__ W2bf,
    const float* __restrict__ b2, float* __restrict__ io) {
    __shared__ unsigned short sHid[MT * DHID];   // bf16 bits, row-major 32 x 1024 (64 KB)

    const int lane = threadIdx.x;
    const int M  = lane & 15;
    const int hi = lane >> 4;
    const int kk = blockIdx.y;
    const int b0 = blockIdx.x * MT;

    // ---- A tiles: ys rows (16 x 12 each) + ones-row at K=12, zero-padded to 16x32
    BF16x16 a0[2];
    #pragma unroll
    for (int t = 0; t < 2; ++t) {
        const float* ysrow = io + (size_t)(b0 + 16 * t + M) * OUTF + kk * WS;
        #pragma unroll
        for (int e = 0; e < 16; ++e) {
            int ke = ((e < 8) ? e : (16 + e - 8)) + hi * 8;
            unsigned short val = 0;
            if (ke < WS)       val = f2bf(ysrow[ke]);
            else if (ke == WS) val = 0x3F80;       // bf16(1.0): bias row selector
            a0[t].u[e] = val;
        }
    }

    // ---- stage 1: 64 iterations, 2 WMMAs per B load
    const uint4* W1p = (const uint4*)(W1bf + (size_t)kk * 64 * 32 * 16);
    for (int n = 0; n < 64; ++n) {
        __builtin_prefetch(&W1p[((n + 1) * 32 + lane) * 2], 0, 3);   // speculative
        BF16x16 bm;
        bm.q[0] = W1p[(n * 32 + lane) * 2 + 0];
        bm.q[1] = W1p[(n * 32 + lane) * 2 + 1];
        #pragma unroll
        for (int t = 0; t < 2; ++t) {
            v8f c = {};
            c = __builtin_amdgcn_wmma_f32_16x16x32_bf16(false, a0[t].v, false, bm.v,
                                                        (short)0, c, false, false);
            #pragma unroll
            for (int r = 0; r < 8; ++r)
                sHid[(16 * t + r + hi * 8) * DHID + n * 16 + M] = f2bf_trunc(c[r]);
        }
    }
    __syncthreads();   // single-wave workgroup: S_NOP + LDS wait

    // ---- stage 2: 32 iterations, A via ds_load_b128, 2 WMMAs per B load
    const uint4* W2p = (const uint4*)(W2bf + (size_t)kk * 32 * 32 * 16);
    v8f acc0 = {}, acc1 = {};
    for (int c2 = 0; c2 < 32; ++c2) {
        BF16x16 bm;
        bm.q[0] = W2p[(c2 * 32 + lane) * 2 + 0];
        bm.q[1] = W2p[(c2 * 32 + lane) * 2 + 1];
        BF16x16 a2;
        const uint4* sp0 = (const uint4*)&sHid[M * DHID + c2 * 32];
        a2.q[0] = sp0[hi];          // shorts [hi*8 .. hi*8+7]
        a2.q[1] = sp0[2 + hi];      // shorts [16+hi*8 .. 16+hi*8+7]
        acc0 = __builtin_amdgcn_wmma_f32_16x16x32_bf16(false, a2.v, false, bm.v,
                                                       (short)0, acc0, false, false);
        const uint4* sp1 = (const uint4*)&sHid[(16 + M) * DHID + c2 * 32];
        a2.q[0] = sp1[hi];
        a2.q[1] = sp1[2 + hi];
        acc1 = __builtin_amdgcn_wmma_f32_16x16x32_bf16(false, a2.v, false, bm.v,
                                                       (short)0, acc1, false, false);
    }

    // ---- epilogue: + b2, overwrite the ys tiles with the result
    if (M < WS) {
        float bias = b2[(size_t)kk * WS + M];
        #pragma unroll
        for (int r = 0; r < 8; ++r) {
            int row = r + hi * 8;
            io[(size_t)(b0 + row) * OUTF + kk * WS + M]      = acc0[r] + bias;
            io[(size_t)(b0 + 16 + row) * OUTF + kk * WS + M] = acc1[r] + bias;
        }
    }
}

// ---------------------------------------------------------------------------
extern "C" void kernel_launch(void* const* d_in, const int* in_sizes, int n_in,
                              void* d_out, int out_size, void* d_ws, size_t ws_size,
                              hipStream_t stream) {
    const float* x    = (const float*)d_in[0];
    const float* encW = (const float*)d_in[1];
    const float* encB = (const float*)d_in[2];
    const float* Wih  = (const float*)d_in[3];
    const float* Whh  = (const float*)d_in[4];
    const float* bih  = (const float*)d_in[5];
    const float* bhh  = (const float*)d_in[6];
    const float* W1   = (const float*)d_in[7];
    const float* b1   = (const float*)d_in[8];
    const float* W2   = (const float*)d_in[9];
    const float* b2   = (const float*)d_in[10];
    float* out = (float*)d_out;

    // workspace layout (all 16B aligned)
    float* ws = (float*)d_ws;
    float*          xs   = ws;                                        // 1,966,080 f32
    float*          pe   = ws + (size_t)BATCH * KSEG * WS;            // 1,920 f32
    unsigned short* W1bf = (unsigned short*)(pe + KSEG * WS);         // 5,242,880 u16
    unsigned short* W2bf = W1bf + (size_t)KSEG * 64 * 32 * 16;        // 2,621,440 u16

    enc_kernel<<<(BATCH * KSEG * WS + 255) / 256, 256, 0, stream>>>(x, encW, encB, xs);
    pe_kernel<<<(KSEG * WS + 255) / 256, 256, 0, stream>>>(pe);
    pack_w1_kernel<<<(KSEG * 64 * 32 + 255) / 256, 256, 0, stream>>>(W1, b1, W1bf);
    pack_w2_kernel<<<(KSEG * 32 * 32 + 255) / 256, 256, 0, stream>>>(W2, W2bf);
    gru_kernel<<<BATCH / 16, 256, 0, stream>>>(xs, pe, Wih, Whh, bih, bhh, out);
    dec_kernel<<<dim3(BATCH / MT, KSEG), 32, 0, stream>>>(W1bf, W2bf, b2, out);
    (void)in_sizes; (void)n_in; (void)out_size; (void)ws_size;
}
